// GraphModel_2173253452180
// MI455X (gfx1250) — matrix operand order repaired and checked by reading, not compiled, not run on hardware
//
#include <hip/hip_runtime.h>
#include <hip/hip_bf16.h>

// ---------------------------------------------------------------- constants
constexpr int NS = 10000;     // source nodes
constexpr int NE = 320000;    // edges
constexpr int DD = 128;       // hidden dim
constexpr int ED = 236;       // edge feature dim
constexpr int AP = 40;        // LDS tile pitch in bf16 elems (16B aligned rows, conflict-reducing)

typedef __attribute__((ext_vector_type(16))) __bf16 v16bf;
typedef __attribute__((ext_vector_type(8)))  __bf16 v8bf;
typedef __attribute__((ext_vector_type(8)))  float  v8f;

__device__ __forceinline__ unsigned pack2(float x, float y) {
    union { __bf16 b[2]; unsigned u; } t;
    t.b[0] = (__bf16)x;
    t.b[1] = (__bf16)y;
    return t.u;
}

// ---------------------------------------------------------------- WMMA core
// lA: row-major [128 rows][AP] bf16, K-step = 32 cols used.
// lB: transposed weight tile [128 n][AP] bf16 (k contiguous).
__device__ __forceinline__ v16bf ld16(const __bf16* p, int off) {
    v8bf a0 = *(const v8bf*)p;
    v8bf a1 = *(const v8bf*)(p + off);
    return __builtin_shufflevector(a0, a1, 0,1,2,3,4,5,6,7,8,9,10,11,12,13,14,15);
}

__device__ __forceinline__ void wmma32(const __bf16* lA, const __bf16* lB,
                                       int w, int l, v8f* acc) {
    const int mm = l & 15, h = l >> 4;
    // A 16x32 bf16: lane half h -> elems 0..7 = K 8h..8h+7, elems 8..15 = K 16+8h..
    v16bf a = ld16(lA + (w * 16 + mm) * AP + 8 * h, 16);
    // Load all 8 B operands into distinct registers so the scheduler can issue
    // every ds_load up front and use partial s_wait_dscnt instead of dscnt==0.
    v16bf b[8];
#pragma unroll
    for (int c = 0; c < 8; ++c)
        b[c] = ld16(lB + (c * 16 + mm) * AP + 8 * h, 16);
#pragma unroll
    for (int c = 0; c < 8; ++c)
        acc[c] = __builtin_amdgcn_wmma_f32_16x16x32_bf16(
            false, a, false, b[c], (short)0, acc[c], false, false);
}

// ---------------------------------------------------------------- staging
// All staging packs 2 consecutive K elems per thread: float2 global load
// (8B-aligned: every lda used is even) + one packed ds_store_b32.
__device__ __forceinline__ void stageA(__bf16* lA, const float* A, int row0,
                                       int M, int lda, int K, int k0, int tid) {
    for (int idx = tid; idx < 128 * 16; idx += 256) {
        int r = idx >> 4, k2 = (idx & 15) * 2;
        int gm = row0 + r, gk = k0 + k2;
        float x = 0.f, y = 0.f;
        if (gm < M) {
            if (gk + 1 < K) {
                float2 v = *(const float2*)(A + (long)gm * lda + gk);
                x = v.x; y = v.y;
            } else if (gk < K) {
                x = A[(long)gm * lda + gk];
            }
        }
        *(unsigned*)&lA[r * AP + k2] = pack2(x, y);
    }
}

__device__ __forceinline__ void stageA_bn(__bf16* lA, const float* A,
                                          const float* bnp, int row0, int M,
                                          int k0, int tid) {
    for (int idx = tid; idx < 128 * 16; idx += 256) {
        int r = idx >> 4, k2 = (idx & 15) * 2;
        int gm = row0 + r, gk = k0 + k2;
        float x = 0.f, y = 0.f;
        if (gm < M) {                       // gk always < 128 here
            float2 v = *(const float2*)(A + (long)gm * DD + gk);
            x = fmaxf(v.x * bnp[2 * gk]     + bnp[2 * gk + 1], 0.f);
            y = fmaxf(v.y * bnp[2 * gk + 2] + bnp[2 * gk + 3], 0.f);
        }
        *(unsigned*)&lA[r * AP + k2] = pack2(x, y);
    }
}

__device__ __forceinline__ void stageA_gather(__bf16* lA, const float* A,
                                              const int* sids, int k0, int tid) {
    for (int idx = tid; idx < 128 * 16; idx += 256) {
        int r = idx >> 4, k2 = (idx & 15) * 2;
        float2 v = *(const float2*)(A + (long)sids[r] * DD + k0 + k2);
        *(unsigned*)&lA[r * AP + k2] = pack2(v.x, v.y);
    }
}

__device__ __forceinline__ void stageB(__bf16* lB, const float* W, int K,
                                       int k0, int tid) {
    // W is [K,128] row-major; store transposed: lB[n][kk]
    for (int idx = tid; idx < 128 * 16; idx += 256) {
        int k2 = (idx >> 7) * 2, n = idx & 127;
        int gk = k0 + k2;
        float x = (gk < K)     ? W[(long)gk * DD + n]       : 0.f;
        float y = (gk + 1 < K) ? W[(long)(gk + 1) * DD + n] : 0.f;
        *(unsigned*)&lB[n * AP + k2] = pack2(x, y);
    }
}

__device__ __forceinline__ void acc_zero(v8f* acc) {
    v8f z = {0.f, 0.f, 0.f, 0.f, 0.f, 0.f, 0.f, 0.f};
#pragma unroll
    for (int c = 0; c < 8; ++c) acc[c] = z;
}

// ---------------------------------------------------------------- kernels
__global__ void k_zero(float* p, long n) {
    long i = (long)blockIdx.x * blockDim.x + threadIdx.x;
    long st = (long)gridDim.x * blockDim.x;
    for (; i < n; i += st) p[i] = 0.f;
}

__global__ __launch_bounds__(256) void k_scatter(const float* pf,
                                                 const int* src_ids, float* agg) {
    int e = blockIdx.x;
    int t = threadIdx.x;
    int sid = src_ids[e];
    if (t < ED) {
        float v = fmaxf(pf[(long)e * ED + t], 0.f);
        atomicAdd(&agg[(long)sid * ED + t], v);
    }
}

// h_i = agg @ We_i + be_i + paths_srcs
__global__ __launch_bounds__(256) void k_gemm_h(const float* agg, const float* We,
                                                const float* be, const float* srcs,
                                                float* h_all) {
    int i = blockIdx.z;
    int row0 = blockIdx.x * 128;
    __shared__ __align__(16) __bf16 lA[128 * AP];
    __shared__ __align__(16) __bf16 lB[128 * AP];
    v8f acc[8]; acc_zero(acc);
    const float* W = We + (long)i * ED * DD;
    for (int ks = 0; ks < 8; ++ks) {
        __syncthreads();
        stageA(lA, agg, row0, NS, ED, ED, ks * 32, threadIdx.x);
        stageB(lB, W, ED, ks * 32, threadIdx.x);
        __syncthreads();
        wmma32(lA, lB, threadIdx.x >> 5, threadIdx.x & 31, acc);
    }
    int l = threadIdx.x & 31, w = threadIdx.x >> 5, h = l >> 4, nl = l & 15;
    float* out = h_all + (long)i * NS * DD;
#pragma unroll
    for (int c = 0; c < 8; ++c)
#pragma unroll
        for (int v = 0; v < 8; ++v) {
            int gm = row0 + w * 16 + v + 8 * h;
            int n = c * 16 + nl;
            if (gm < NS)
                out[(long)gm * DD + n] = acc[c][v] + be[i * DD + n] + srcs[(long)gm * DD + n];
        }
}

// t_i = h_i @ W1_i ; accumulate BN batch stats
__global__ __launch_bounds__(256) void k_gemm_t(const float* h_all, const float* W1,
                                                float* t_all, float* ssum, float* ssq) {
    int i = blockIdx.z;
    int row0 = blockIdx.x * 128;
    __shared__ __align__(16) __bf16 lA[128 * AP];
    __shared__ __align__(16) __bf16 lB[128 * AP];
    v8f acc[8]; acc_zero(acc);
    const float* A = h_all + (long)i * NS * DD;
    const float* W = W1 + (long)i * DD * DD;
    for (int ks = 0; ks < 4; ++ks) {
        __syncthreads();
        stageA(lA, A, row0, NS, DD, DD, ks * 32, threadIdx.x);
        stageB(lB, W, DD, ks * 32, threadIdx.x);
        __syncthreads();
        wmma32(lA, lB, threadIdx.x >> 5, threadIdx.x & 31, acc);
    }
    int l = threadIdx.x & 31, w = threadIdx.x >> 5, h = l >> 4, nl = l & 15;
    float* out = t_all + (long)i * NS * DD;
#pragma unroll
    for (int c = 0; c < 8; ++c) {
        int n = c * 16 + nl;
        float s = 0.f, s2 = 0.f;
#pragma unroll
        for (int v = 0; v < 8; ++v) {
            int gm = row0 + w * 16 + v + 8 * h;
            if (gm < NS) {
                float x = acc[c][v];
                out[(long)gm * DD + n] = x;
                s += x; s2 += x * x;
            }
        }
        atomicAdd(&ssum[i * DD + n], s);
        atomicAdd(&ssq[i * DD + n], s2);
    }
}

__global__ void k_bnfin(const float* sum, const float* sq, const float* g,
                        const float* b, float invN, float* bnp, int n) {
    int t = blockIdx.x * blockDim.x + threadIdx.x;
    if (t < n) {
        float m = sum[t] * invN;
        float var = sq[t] * invN - m * m;
        float sc = g[t] * rsqrtf(var + 1e-5f);
        bnp[2 * t] = sc;
        bnp[2 * t + 1] = b[t] - m * sc;
    }
}

// hs_i = relu(bn(t_i)) @ W2_i, written into hcat column block i
__global__ __launch_bounds__(256) void k_gemm_hs(const float* t_all, const float* bnp,
                                                 const float* W2, float* hcat) {
    int i = blockIdx.z;
    int row0 = blockIdx.x * 128;
    __shared__ __align__(16) __bf16 lA[128 * AP];
    __shared__ __align__(16) __bf16 lB[128 * AP];
    v8f acc[8]; acc_zero(acc);
    const float* A = t_all + (long)i * NS * DD;
    const float* W = W2 + (long)i * DD * DD;
    const float* bp = bnp + i * 2 * DD;
    for (int ks = 0; ks < 4; ++ks) {
        __syncthreads();
        stageA_bn(lA, A, bp, row0, NS, ks * 32, threadIdx.x);
        stageB(lB, W, DD, ks * 32, threadIdx.x);
        __syncthreads();
        wmma32(lA, lB, threadIdx.x >> 5, threadIdx.x & 31, acc);
    }
    int l = threadIdx.x & 31, w = threadIdx.x >> 5, h = l >> 4, nl = l & 15;
#pragma unroll
    for (int c = 0; c < 8; ++c)
#pragma unroll
        for (int v = 0; v < 8; ++v) {
            int gm = row0 + w * 16 + v + 8 * h;
            int n = c * 16 + nl;
            if (gm < NS) hcat[(long)gm * (3 * DD) + i * DD + n] = acc[c][v];
        }
}

// u = relu(hcat @ lin1_W + lin1_b)
__global__ __launch_bounds__(256) void k_gemm_u(const float* hcat, const float* W,
                                                const float* bias, float* u) {
    int row0 = blockIdx.x * 128;
    __shared__ __align__(16) __bf16 lA[128 * AP];
    __shared__ __align__(16) __bf16 lB[128 * AP];
    v8f acc[8]; acc_zero(acc);
    for (int ks = 0; ks < 12; ++ks) {
        __syncthreads();
        stageA(lA, hcat, row0, NS, 3 * DD, 3 * DD, ks * 32, threadIdx.x);
        stageB(lB, W, 3 * DD, ks * 32, threadIdx.x);
        __syncthreads();
        wmma32(lA, lB, threadIdx.x >> 5, threadIdx.x & 31, acc);
    }
    int l = threadIdx.x & 31, w = threadIdx.x >> 5, h = l >> 4, nl = l & 15;
#pragma unroll
    for (int c = 0; c < 8; ++c)
#pragma unroll
        for (int v = 0; v < 8; ++v) {
            int gm = row0 + w * 16 + v + 8 * h;
            int n = c * 16 + nl;
            if (gm < NS) u[(long)gm * DD + n] = fmaxf(acc[c][v] + bias[n], 0.f);
        }
}

// node_emb = u @ lin2_W + lin2_b   (writes d_out)
__global__ __launch_bounds__(256) void k_gemm_node(const float* u, const float* W,
                                                   const float* bias, float* out) {
    int row0 = blockIdx.x * 128;
    __shared__ __align__(16) __bf16 lA[128 * AP];
    __shared__ __align__(16) __bf16 lB[128 * AP];
    v8f acc[8]; acc_zero(acc);
    for (int ks = 0; ks < 4; ++ks) {
        __syncthreads();
        stageA(lA, u, row0, NS, DD, DD, ks * 32, threadIdx.x);
        stageB(lB, W, DD, ks * 32, threadIdx.x);
        __syncthreads();
        wmma32(lA, lB, threadIdx.x >> 5, threadIdx.x & 31, acc);
    }
    int l = threadIdx.x & 31, w = threadIdx.x >> 5, h = l >> 4, nl = l & 15;
#pragma unroll
    for (int c = 0; c < 8; ++c)
#pragma unroll
        for (int v = 0; v < 8; ++v) {
            int gm = row0 + w * 16 + v + 8 * h;
            int n = c * 16 + nl;
            if (gm < NS) out[(long)gm * DD + n] = acc[c][v] + bias[n];
        }
}

// hpath = node_emb[src_ids] @ pe_src_W + path_feats @ pe_tgt_W + biases; BN stats
__global__ __launch_bounds__(256) void k_gemm_path(const float* node_emb, const float* pf,
                                                   const int* src_ids, const float* Wsrc,
                                                   const float* Wtgt, const float* bsrc,
                                                   const float* btgt, float* hpath,
                                                   float* psum, float* psq) {
    int row0 = blockIdx.x * 128;
    __shared__ __align__(16) __bf16 lA[128 * AP];
    __shared__ __align__(16) __bf16 lB[128 * AP];
    __shared__ int sids[128];
    if (threadIdx.x < 128) sids[threadIdx.x] = src_ids[row0 + threadIdx.x];
    v8f acc[8]; acc_zero(acc);
    for (int ks = 0; ks < 4; ++ks) {            // phase 1: gathered node embeddings
        __syncthreads();
        stageA_gather(lA, node_emb, sids, ks * 32, threadIdx.x);
        stageB(lB, Wsrc, DD, ks * 32, threadIdx.x);
        __syncthreads();
        wmma32(lA, lB, threadIdx.x >> 5, threadIdx.x & 31, acc);
    }
    for (int ks = 0; ks < 8; ++ks) {            // phase 2: path features (K=236 pad 256)
        __syncthreads();
        stageA(lA, pf, row0, NE, ED, ED, ks * 32, threadIdx.x);
        stageB(lB, Wtgt, ED, ks * 32, threadIdx.x);
        __syncthreads();
        wmma32(lA, lB, threadIdx.x >> 5, threadIdx.x & 31, acc);
    }
    int l = threadIdx.x & 31, w = threadIdx.x >> 5, h = l >> 4, nl = l & 15;
#pragma unroll
    for (int c = 0; c < 8; ++c) {
        int n = c * 16 + nl;
        float bb = bsrc[n] + btgt[n];
        float s = 0.f, s2 = 0.f;
#pragma unroll
        for (int v = 0; v < 8; ++v) {
            int gm = row0 + w * 16 + v + 8 * h;   // always < NE (2500*128 exact)
            float x = acc[c][v] + bb;
            hpath[(long)gm * DD + n] = x;
            s += x; s2 += x * x;
        }
        atomicAdd(&psum[n], s);
        atomicAdd(&psq[n], s2);
    }
}

// scores = relu(bn(hpath)) . score_w
__global__ __launch_bounds__(256) void k_score(const float* hpath, const float* bnp,
                                               const float* score_w, float* scores) {
    __shared__ float sc[DD], sh[DD], sw[DD];
    if (threadIdx.x < DD) {
        sc[threadIdx.x] = bnp[2 * threadIdx.x];
        sh[threadIdx.x] = bnp[2 * threadIdx.x + 1];
        sw[threadIdx.x] = score_w[threadIdx.x];
    }
    __syncthreads();
    long e = (long)blockIdx.x * blockDim.x + threadIdx.x;
    long st = (long)gridDim.x * blockDim.x;
    for (; e < NE; e += st) {
        const float4* row = (const float4*)(hpath + e * DD);
        float s = 0.f;
#pragma unroll 8
        for (int c = 0; c < 32; ++c) {
            float4 v = row[c];
            int n = 4 * c;
            s += fmaxf(v.x * sc[n]     + sh[n],     0.f) * sw[n];
            s += fmaxf(v.y * sc[n + 1] + sh[n + 1], 0.f) * sw[n + 1];
            s += fmaxf(v.z * sc[n + 2] + sh[n + 2], 0.f) * sw[n + 2];
            s += fmaxf(v.w * sc[n + 3] + sh[n + 3], 0.f) * sw[n + 3];
        }
        scores[e] = s;
    }
}

__device__ __forceinline__ void top6_insert(float* v, int* id, float x, int xi) {
    if (!(x > v[5])) return;
    int j = 5;
    while (j > 0 && x > v[j - 1]) { v[j] = v[j - 1]; id[j] = id[j - 1]; --j; }
    v[j] = x; id[j] = xi;
}

__global__ __launch_bounds__(1024) void k_topk(const float* scores, float* out) {
    __shared__ float sv[1024 * 6];
    __shared__ int   si[1024 * 6];
    float bv[6]; int bi[6];
#pragma unroll
    for (int j = 0; j < 6; ++j) { bv[j] = -3.4e38f; bi[j] = 0; }
    for (int e = threadIdx.x; e < NE; e += 1024)
        top6_insert(bv, bi, scores[e], e);
#pragma unroll
    for (int j = 0; j < 6; ++j) {
        sv[threadIdx.x * 6 + j] = bv[j];
        si[threadIdx.x * 6 + j] = bi[j];
    }
    __syncthreads();
    if (threadIdx.x == 0) {
        float gv[6]; int gi[6];
#pragma unroll
        for (int j = 0; j < 6; ++j) { gv[j] = -3.4e38f; gi[j] = 0; }
        for (int k = 0; k < 1024 * 6; ++k) top6_insert(gv, gi, sv[k], si[k]);
#pragma unroll
        for (int j = 0; j < 6; ++j) {
            out[j] = gv[j];
            out[6 + j] = (float)gi[j];
        }
    }
}

// ---------------------------------------------------------------- launch
extern "C" void kernel_launch(void* const* d_in, const int* in_sizes, int n_in,
                              void* d_out, int out_size, void* d_ws, size_t ws_size,
                              hipStream_t stream) {
    (void)in_sizes; (void)n_in; (void)out_size; (void)ws_size;
    const float* paths_srcs  = (const float*)d_in[0];
    const float* path_feats  = (const float*)d_in[1];
    const int*   src_ids     = (const int*)d_in[2];
    const float* conv_edge_W = (const float*)d_in[3];
    const float* conv_edge_b = (const float*)d_in[4];
    const float* conv_W1     = (const float*)d_in[5];
    const float* conv_bn_g   = (const float*)d_in[6];
    const float* conv_bn_b   = (const float*)d_in[7];
    const float* conv_W2     = (const float*)d_in[8];
    const float* lin1_W      = (const float*)d_in[9];
    const float* lin1_b      = (const float*)d_in[10];
    const float* lin2_W      = (const float*)d_in[11];
    const float* lin2_b      = (const float*)d_in[12];
    const float* pe_src_W    = (const float*)d_in[13];
    const float* pe_src_b    = (const float*)d_in[14];
    const float* pe_tgt_W    = (const float*)d_in[15];
    const float* pe_tgt_b    = (const float*)d_in[16];
    const float* pe_bn_g     = (const float*)d_in[17];
    const float* pe_bn_b     = (const float*)d_in[18];
    const float* score_w     = (const float*)d_in[19];

    float* ws = (float*)d_ws;
    // workspace layout (floats)
    float* agg      = ws;                          // 2,360,000
    float* conv_sum = ws + 2360000;                // 384
    float* conv_sq  = conv_sum + 384;              // 384
    float* pe_sum   = conv_sq + 384;               // 128
    float* pe_sq    = pe_sum + 128;                // 128
    float* bnp_conv = pe_sq + 128;                 // 768
    float* bnp_pe   = bnp_conv + 768;              // 256
    float* h_all    = bnp_pe + 256;                // 3*1,280,000
    float* t_all    = h_all + 3840000;             // 3*1,280,000
    float* hcat     = t_all + 3840000;             // 3,840,000
    float* u        = hcat + 3840000;              // 1,280,000
    float* scores   = u + 1280000;                 // 320,000
    float* hpath    = scores + 320000;             // 40,960,000

    float* node_emb = (float*)d_out;               // [NS*DD]
    float* topout   = node_emb + (long)NS * DD;    // 12 floats (6 vals, 6 idx)

    // zero agg + stats accumulators (ws is poisoned, not re-zeroed by harness)
    k_zero<<<4096, 256, 0, stream>>>(agg, 2360000L + 1024L);

    k_scatter<<<NE, 256, 0, stream>>>(path_feats, src_ids, agg);

    dim3 g3(79, 1, 3);
    k_gemm_h<<<g3, 256, 0, stream>>>(agg, conv_edge_W, conv_edge_b, paths_srcs, h_all);
    k_gemm_t<<<g3, 256, 0, stream>>>(h_all, conv_W1, t_all, conv_sum, conv_sq);
    k_bnfin<<<2, 256, 0, stream>>>(conv_sum, conv_sq, conv_bn_g, conv_bn_b,
                                   1.f / NS, bnp_conv, 384);
    k_gemm_hs<<<g3, 256, 0, stream>>>(t_all, bnp_conv, conv_W2, hcat);
    k_gemm_u<<<79, 256, 0, stream>>>(hcat, lin1_W, lin1_b, u);
    k_gemm_node<<<79, 256, 0, stream>>>(u, lin2_W, lin2_b, node_emb);

    k_gemm_path<<<2500, 256, 0, stream>>>(node_emb, path_feats, src_ids,
                                          pe_src_W, pe_tgt_W, pe_src_b, pe_tgt_b,
                                          hpath, pe_sum, pe_sq);
    k_bnfin<<<1, 128, 0, stream>>>(pe_sum, pe_sq, pe_bn_g, pe_bn_b,
                                   1.f / NE, bnp_pe, 128);
    k_score<<<1024, 256, 0, stream>>>(hpath, bnp_pe, score_w, scores);
    k_topk<<<1, 1024, 0, stream>>>(scores, topout);
}
